// SortSplitMLP_63660005262007
// MI455X (gfx1250) — matrix operand order repaired and checked by reading, not compiled, not run on hardware
//
#include <hip/hip_runtime.h>
#include <hip/hip_bf16.h>

// ---------------------------------------------------------------------------
// Problem constants (fixed by the reference)
// ---------------------------------------------------------------------------
#define NTOK   16384
#define HDIM   2048
#define NEXP   8
#define IPE    1024            // expert intermediate size
#define CHUNK  2048            // tokens per expert = NTOK / NEXP
#define TWO_I  2048            // 2 * IPE
#define KB     128             // K-depth of the LDS-staged A tile
#define LDS_STRIDE 136         // 128 + 8 bf16 pad -> 272B row stride (bank spread)

typedef __bf16 bf16;
typedef __attribute__((ext_vector_type(8)))  __bf16 v8bf;
typedef __attribute__((ext_vector_type(16))) __bf16 v16bf;
typedef __attribute__((ext_vector_type(8)))  float  v8f;
typedef __attribute__((ext_vector_type(4)))  int    v4i;

// ---------------------------------------------------------------------------
// fp32 -> bf16 round-to-nearest-even (bit trick)
// ---------------------------------------------------------------------------
static __device__ __forceinline__ unsigned short f2bf(float f) {
  union { float f; unsigned u; } v; v.f = f;
  unsigned u = v.u;
  u += 0x7FFFu + ((u >> 16) & 1u);
  return (unsigned short)(u >> 16);
}

static __device__ __forceinline__ unsigned pack2(float x, float y) {
  return (unsigned)f2bf(x) | ((unsigned)f2bf(y) << 16);
}

// Load a 16-element bf16 fragment as two 16B chunks (chunk2 at +off2 elems).
static __device__ __forceinline__ v16bf ld2x8(const bf16* p, int off2) {
  v8bf lo = *(const v8bf*)p;
  v8bf hi = *(const v8bf*)(p + off2);
  return __builtin_shufflevector(lo, hi, 0, 1, 2, 3, 4, 5, 6, 7,
                                         8, 9, 10, 11, 12, 13, 14, 15);
}

static __device__ __forceinline__ v8f wmma_bf16(v16bf a, v16bf b, v8f c) {
  return __builtin_amdgcn_wmma_f32_16x16x32_bf16(
      /*neg_a=*/false, a, /*neg_b=*/false, b,
      /*c_mod=*/(short)0, c, /*reuse_a=*/false, /*reuse_b=*/false);
}

// ---------------------------------------------------------------------------
// 16B global -> LDS copy. Prefers the CDNA5 async path (ASYNCcnt-tracked
// GLOBAL_LOAD_ASYNC_TO_LDS_B128); falls back to a sync load + ds_store.
// ---------------------------------------------------------------------------
#if __has_builtin(__builtin_amdgcn_global_load_async_to_lds_b128)
#define HAVE_ASYNC_LDS 1
#else
#define HAVE_ASYNC_LDS 0
#endif

#if HAVE_ASYNC_LDS
typedef __attribute__((address_space(1))) v4i* v4i_gp;   // ptr to global int4
typedef __attribute__((address_space(3))) v4i* v4i_lp;   // ptr to LDS int4
#endif

static __device__ __forceinline__ void cp16_g2l(const bf16* gsrc,
                                                unsigned short* ldst) {
#if HAVE_ASYNC_LDS
  __builtin_amdgcn_global_load_async_to_lds_b128(
      (v4i_gp)(v4i*)(bf16*)gsrc, (v4i_lp)(v4i*)ldst, 0, 0);
#else
  *(v8bf*)ldst = *(const v8bf*)gsrc;
#endif
}

static __device__ __forceinline__ void wait_async() {
#if HAVE_ASYNC_LDS
  asm volatile("s_wait_asynccnt 0" ::: "memory");
#endif
}

// Issue one 64 x KB A-tile (1024 x 16B chunks; 4 per thread, coalesced rows).
static __device__ __forceinline__ void issue_a_tile(
    const bf16* A, int lda, int k0, unsigned short (*dst)[LDS_STRIDE]) {
#pragma unroll
  for (int l = 0; l < 4; ++l) {
    const int q = l * 256 + threadIdx.x;     // 0..1023
    const int r = q >> 4;                    // 0..63
    const int c = (q & 15) * 8;              // 0..120, 8-elem chunks
    cp16_g2l(A + (size_t)r * lda + k0 + c, &dst[r][c]);
  }
}

// ---------------------------------------------------------------------------
// Kernel 0: gather tokens in sorted order, convert fp32 -> bf16
// ---------------------------------------------------------------------------
__global__ __launch_bounds__(256) void k_gather_cvt(
    const float* __restrict__ hs, const int* __restrict__ sidx,
    unsigned short* __restrict__ Xs) {
  const int row = blockIdx.x;
  const int t = threadIdx.x;                 // 256 threads * 8 floats = 2048
  const int src = sidx[row];
  const float4* ip = (const float4*)(hs + (size_t)src * HDIM) + (size_t)t * 2;
  float4 a = ip[0], b = ip[1];
  uint4 u;
  u.x = pack2(a.x, a.y);
  u.y = pack2(a.z, a.w);
  u.z = pack2(b.x, b.y);
  u.w = pack2(b.z, b.w);
  ((uint4*)(Xs + (size_t)row * HDIM))[t] = u;
}

// ---------------------------------------------------------------------------
// Kernel 1: per-expert transpose + convert fp32 [R, C] -> bf16 [C, R]
// ---------------------------------------------------------------------------
__global__ __launch_bounds__(256) void k_transpose_cvt(
    const float* __restrict__ in, unsigned short* __restrict__ out,
    int R, int C) {
  __shared__ float tile[32][33];
  const int cb = blockIdx.x * 32;
  const int rb = blockIdx.y * 32;
  const int tx = threadIdx.x;                // 0..31
  const int ty = threadIdx.y;                // 0..7
  const float* ip = in + (size_t)blockIdx.z * R * C;
  unsigned short* op = out + (size_t)blockIdx.z * R * C;
#pragma unroll
  for (int i = 0; i < 32; i += 8)
    tile[ty + i][tx] = ip[(size_t)(rb + ty + i) * C + cb + tx];
  __syncthreads();
#pragma unroll
  for (int i = 0; i < 32; i += 8)
    op[(size_t)(cb + ty + i) * R + rb + tx] = f2bf(tile[tx][ty + i]);
}

// ---------------------------------------------------------------------------
// Kernel 2: GEMM1 + SiLU, A tile staged in LDS via async copies.
// Wave tile: M=64, N=32 over the gate half AND matching up half.
// Block = 8 waves: shared A (M=64 x K=128 double-buffered), N=256 gate cols.
// Grid: (IPE/256, CHUNK/64, NEXP) = (4, 32, 8)
// ---------------------------------------------------------------------------
__global__ __launch_bounds__(256) void k_gemm1_silu(
    const bf16* __restrict__ Xs, const bf16* __restrict__ WguT,
    unsigned short* __restrict__ Act) {
  __shared__ __align__(16) unsigned short As[2][64][LDS_STRIDE];

  const int lane = threadIdx.x & 31;
  const int wave = threadIdx.x >> 5;
  const int l16 = lane & 15;
  const int half = lane >> 4;

  const int e  = blockIdx.z;
  const int m0 = blockIdx.y * 64;
  const int n0 = blockIdx.x * 256 + wave * 32;   // gate column base

  const bf16* A  = Xs   + ((size_t)e * CHUNK + m0) * HDIM;
  const bf16* Bg = WguT + (size_t)e * TWO_I * HDIM + (size_t)n0 * HDIM;
  const bf16* Bu = Bg   + (size_t)IPE * HDIM;

  v8f cg[4][2] = {};
  v8f cu[4][2] = {};

  issue_a_tile(A, HDIM, 0, As[0]);
  int p = 0;
  for (int kb = 0; kb < HDIM; kb += KB) {
    wait_async();
    __syncthreads();                       // As[p] ready for all waves
    if (kb + KB < HDIM) issue_a_tile(A, HDIM, kb + KB, As[p ^ 1]);
#pragma unroll
    for (int kk = 0; kk < KB; kk += 32) {
      v16bf a[4], bg[2], bu[2];
#pragma unroll
      for (int i = 0; i < 4; ++i)
        a[i] = ld2x8((const bf16*)&As[p][16 * i + l16][kk + 16 * half], 8);
#pragma unroll
      for (int j = 0; j < 2; ++j) {
        const int k0 = kb + kk + 16 * half;
        bg[j] = ld2x8(Bg + (size_t)(16 * j + l16) * HDIM + k0, 8);
        bu[j] = ld2x8(Bu + (size_t)(16 * j + l16) * HDIM + k0, 8);
      }
#pragma unroll
      for (int i = 0; i < 4; ++i)
#pragma unroll
        for (int j = 0; j < 2; ++j) {
          cg[i][j] = wmma_bf16(a[i], bg[j], cg[i][j]);
          cu[i][j] = wmma_bf16(a[i], bu[j], cu[i][j]);
        }
    }
    p ^= 1;
  }

  // Epilogue: silu(gate) * up -> bf16  (fast rcp: bf16-store precision)
  unsigned short* actp = Act + ((size_t)e * CHUNK + m0) * IPE;
#pragma unroll
  for (int i = 0; i < 4; ++i)
#pragma unroll
    for (int j = 0; j < 2; ++j)
#pragma unroll
      for (int v = 0; v < 8; ++v) {
        const int m = 16 * i + v + 8 * half;
        const int n = n0 + 16 * j + l16;
        float g = cg[i][j][v];
        float u = cu[i][j][v];
        float s = g * __builtin_amdgcn_rcpf(1.0f + __expf(-g));
        actp[(size_t)m * IPE + n] = f2bf(s * u);
      }
}

// ---------------------------------------------------------------------------
// Kernel 3: GEMM2 + scatter, A (activations) staged in LDS via async copies.
// Wave tile: M=64, N=64. Block = 8 waves: shared A, N=512.
// Grid: (HDIM/512, CHUNK/64, NEXP) = (4, 32, 8)
// ---------------------------------------------------------------------------
__global__ __launch_bounds__(256) void k_gemm2_scatter(
    const bf16* __restrict__ Act, const bf16* __restrict__ WdT,
    const int* __restrict__ sidx, float* __restrict__ out) {
  __shared__ __align__(16) unsigned short As[2][64][LDS_STRIDE];

  const int lane = threadIdx.x & 31;
  const int wave = threadIdx.x >> 5;
  const int l16 = lane & 15;
  const int half = lane >> 4;

  const int e  = blockIdx.z;
  const int m0 = blockIdx.y * 64;
  const int n0 = blockIdx.x * 512 + wave * 64;

  const bf16* A  = Act + ((size_t)e * CHUNK + m0) * IPE;
  const bf16* BT = WdT + (size_t)e * HDIM * IPE + (size_t)n0 * IPE;

  v8f c[4][4] = {};

  issue_a_tile(A, IPE, 0, As[0]);
  int p = 0;
  for (int kb = 0; kb < IPE; kb += KB) {
    wait_async();
    __syncthreads();
    if (kb + KB < IPE) issue_a_tile(A, IPE, kb + KB, As[p ^ 1]);
#pragma unroll
    for (int kk = 0; kk < KB; kk += 32) {
      v16bf a[4], b[4];
#pragma unroll
      for (int i = 0; i < 4; ++i)
        a[i] = ld2x8((const bf16*)&As[p][16 * i + l16][kk + 16 * half], 8);
#pragma unroll
      for (int j = 0; j < 4; ++j)
        b[j] = ld2x8(BT + (size_t)(16 * j + l16) * IPE + kb + kk + 16 * half, 8);
#pragma unroll
      for (int i = 0; i < 4; ++i)
#pragma unroll
        for (int j = 0; j < 4; ++j)
          c[i][j] = wmma_bf16(a[i], b[j], c[i][j]);
    }
    p ^= 1;
  }

  // Epilogue: scatter rows through sort_idx
#pragma unroll
  for (int i = 0; i < 4; ++i)
#pragma unroll
    for (int v = 0; v < 8; ++v) {
      const int mrow = e * CHUNK + m0 + 16 * i + v + 8 * half;
      const int dst = sidx[mrow];
      float* op = out + (size_t)dst * HDIM;
#pragma unroll
      for (int j = 0; j < 4; ++j)
        op[n0 + 16 * j + l16] = c[i][j][v];
    }
}

// ---------------------------------------------------------------------------
// Host-side launcher
// ---------------------------------------------------------------------------
extern "C" void kernel_launch(void* const* d_in, const int* in_sizes, int n_in,
                              void* d_out, int out_size, void* d_ws, size_t ws_size,
                              hipStream_t stream) {
  const float* hs   = (const float*)d_in[0];   // [NTOK, HDIM] fp32
  const int*   sidx = (const int*)d_in[1];     // [NTOK] int
  const float* wgu  = (const float*)d_in[2];   // [E, H, 2I] fp32
  const float* wd   = (const float*)d_in[3];   // [E, I, H] fp32
  float* out = (float*)d_out;                  // [NTOK, HDIM] fp32

  // Workspace layout (bytes):
  //   Xs   bf16 [NTOK, H]     : 64 MB @ 0
  //   WguT bf16 [E, 2I, H]    : 64 MB @ 64MB
  //   WdT  bf16 [E, H, I]     : 32 MB @ 128MB
  //   Act  bf16 [NTOK, I]     : 32 MB @ 160MB
  char* base = (char*)d_ws;
  unsigned short* Xs   = (unsigned short*)(base);
  unsigned short* WguT = (unsigned short*)(base + (size_t)64 * 1024 * 1024);
  unsigned short* WdT  = (unsigned short*)(base + (size_t)128 * 1024 * 1024);
  unsigned short* Act  = (unsigned short*)(base + (size_t)160 * 1024 * 1024);

  // 0) gather + fp32->bf16
  k_gather_cvt<<<NTOK, 256, 0, stream>>>(hs, sidx, Xs);

  // 1) weight transposes (fp32 [R,C] -> bf16 [C,R] per expert)
  k_transpose_cvt<<<dim3(TWO_I / 32, HDIM / 32, NEXP), dim3(32, 8), 0, stream>>>(
      wgu, WguT, HDIM, TWO_I);                 // -> [E, 2I, H]
  k_transpose_cvt<<<dim3(HDIM / 32, IPE / 32, NEXP), dim3(32, 8), 0, stream>>>(
      wd, WdT, IPE, HDIM);                     // -> [E, H, I]

  // 2) GEMM1 + SiLU
  k_gemm1_silu<<<dim3(IPE / 256, CHUNK / 64, NEXP), 256, 0, stream>>>(
      (const bf16*)Xs, (const bf16*)WguT, Act);

  // 3) GEMM2 + scatter
  k_gemm2_scatter<<<dim3(HDIM / 512, CHUNK / 64, NEXP), 256, 0, stream>>>(
      (const bf16*)Act, (const bf16*)WdT, sidx, out);
}
